// KeyValueBottleneck_51049981280548
// MI455X (gfx1250) — matrix operand order repaired and checked by reading, not compiled, not run on hardware
//
#include <hip/hip_runtime.h>
#include <stdint.h>

// ---------------- types for WMMA ----------------
typedef __attribute__((ext_vector_type(16))) __bf16 v16bf;
typedef __attribute__((ext_vector_type(8)))  float  v8f;
typedef __attribute__((ext_vector_type(4)))  unsigned int v4u;

union U16 { v16bf v; v4u u[2]; };

// problem constants
#define BATCH 16384
#define NKEY  4096
#define KDIM  1024
#define VDIM  1024
#define NOUT  10

// GEMM tiling
#define KSLAB      64                 // K per LDS slab
#define LDS_STRIDE 72                 // bf16 elements per column row (64 + 8 pad)
#define BUF_ELEMS  (128 * LDS_STRIDE) // one buffer: 9216 bf16 = 18,432 B
#define NSLAB      (KDIM / KSLAB)     // 16 slabs per column group
#define NITER      ((NKEY / 128) * NSLAB) // 512 pipelined iterations

// raw LDS byte offset of a __shared__ pointer (generic -> addrspace(3))
__device__ __forceinline__ unsigned lds_addr_of(const void* p) {
    return (unsigned)(unsigned long long)
        (__attribute__((address_space(3))) const void*)p;
}

// ---------------- fp32 -> bf16 (RTNE) conversion ----------------
__device__ __forceinline__ unsigned int f2bf(float f) {
    union { float f; unsigned int u; } c; c.f = f;
    unsigned int u = c.u;
    return (u + 0x7FFFu + ((u >> 16) & 1u)) >> 16;   // bf16 bits in low 16
}

__global__ void cvt_f32_to_bf16(const float* __restrict__ src,
                                unsigned short* __restrict__ dst, int n8) {
    int i = blockIdx.x * blockDim.x + threadIdx.x;
    if (i >= n8) return;
    const float4* s4 = (const float4*)src;
    float4 a = s4[i * 2 + 0];
    float4 b = s4[i * 2 + 1];
    v4u o;
    o.x = f2bf(a.x) | (f2bf(a.y) << 16);
    o.y = f2bf(a.z) | (f2bf(a.w) << 16);
    o.z = f2bf(b.x) | (f2bf(b.y) << 16);
    o.w = f2bf(b.z) | (f2bf(b.w) << 16);
    ((v4u*)dst)[i] = o;                               // 16B store
}

// ---------------- fused sims GEMM + argmax ----------------
// grid: 128 blocks x 256 threads (8 waves). Block covers 128 rows of x.
// Flattened pipeline over (col-group, k-slab): the next 128x64 bf16 key slab
// is DMA'd into the alternate LDS buffer via global_load_async_to_lds_b128
// while the current slab's 16 B tiles + 2 A fragments are batch-loaded into
// registers and fed to 16 back-to-back WMMAs (sched_barrier pins the split).
__global__ void __launch_bounds__(256)
kv_gemm_argmax(const unsigned short* __restrict__ xb,
               const unsigned short* __restrict__ kb,
               int* __restrict__ nearest) {
    __shared__ __align__(16) unsigned short sk[2 * BUF_ELEMS];   // 36,864 B

    const int tid  = threadIdx.x;
    const int lane = tid & 31;
    const int wave = tid >> 5;
    const int l15  = lane & 15;
    const int hl   = lane >> 4;                       // 0: lanes 0-15, 1: lanes 16-31

    const int rowbase = blockIdx.x * 128 + wave * 16;
    const unsigned short* aptr = xb + (size_t)(rowbase + l15) * KDIM;

    // per-thread invariant staging offsets: 1024 16B-chunks / 256 threads = 4
    int      gb[4];   // global element offset within a (col-group, slab)
    unsigned lo[4];   // LDS byte offset within one buffer
#pragma unroll
    for (int j = 0; j < 4; ++j) {
        int c   = tid + j * 256;                      // 0..1023
        int col = c >> 3;                             // 0..127
        int ch  = c & 7;                              // 16B chunk within 64-K row
        gb[j] = col * KDIM + ch * 8;
        lo[j] = (unsigned)(col * (LDS_STRIDE * 2) + ch * 16);
    }
    const unsigned skbase = lds_addr_of(sk);

    float best[8];
    int   bidx[8];
#pragma unroll
    for (int r = 0; r < 8; ++r) { best[r] = -3.402823466e38f; bidx[r] = 0; }

    v8f acc[8];

    // ---- stage helper: async-DMA one 128x64 key slab into LDS buffer i&1 ----
    auto stage = [&](int i) {
        const unsigned short* g =
            kb + (size_t)((i >> 4) * (128 * KDIM) + (i & 15) * KSLAB);
        const unsigned lbase = skbase + (unsigned)(i & 1) * (BUF_ELEMS * 2);
#pragma unroll
        for (int j = 0; j < 4; ++j) {
            const unsigned short* ga = g + gb[j];
            unsigned la = lbase + lo[j];
            asm volatile("global_load_async_to_lds_b128 %0, %1, off"
                         :: "v"(la), "v"(ga) : "memory");
        }
    };

    stage(0);

#pragma unroll 1
    for (int i = 0; i < NITER; ++i) {
        const int ks = i & (NSLAB - 1);

        if (ks == 0) {
#pragma unroll
            for (int t = 0; t < 8; ++t) acc[t] = 0.f;
        }

        // stage(i) complete locally, then globally; also fences compute(i-1)
        asm volatile("s_wait_asynccnt 0x0" ::: "memory");
        __syncthreads();

        if (i + 1 < NITER) stage(i + 1);              // overlaps compute below

        const unsigned short* buf = sk + (i & 1) * BUF_ELEMS;

        // ---- batch-load the whole slab's operands into registers ----
        U16 A2[2];
        U16 Bm[2][8];
#pragma unroll
        for (int s = 0; s < 2; ++s) {
            const int k = ks * KSLAB + s * 32;
            A2[s].u[0] = *(const v4u*)(aptr + k + hl * 16);
            A2[s].u[1] = *(const v4u*)(aptr + k + hl * 16 + 8);
        }
#pragma unroll
        for (int s = 0; s < 2; ++s) {
#pragma unroll
            for (int t = 0; t < 8; ++t) {
                const unsigned short* bsrc =
                    buf + (t * 16 + l15) * LDS_STRIDE + s * 32 + hl * 16;
                Bm[s][t].u[0] = ((const v4u*)bsrc)[0];
                Bm[s][t].u[1] = ((const v4u*)bsrc)[1];
            }
        }
        // keep all loads above, all matrix ops below: one wait, 16 WMMAs
        __builtin_amdgcn_sched_barrier(0);

#pragma unroll
        for (int s = 0; s < 2; ++s) {
#pragma unroll
            for (int t = 0; t < 8; ++t) {
                acc[t] = __builtin_amdgcn_wmma_f32_16x16x32_bf16(
                    false, A2[s].v, false, Bm[s][t].v, (short)0, acc[t],
                    false, false);
            }
        }

        if (ks == NSLAB - 1) {                        // full K done: argmax
            const int col0 = (i >> 4) * 128;
#pragma unroll
            for (int t = 0; t < 8; ++t) {
                const int col = col0 + t * 16 + l15;
#pragma unroll
                for (int r = 0; r < 8; ++r) {
                    float v = acc[t][r];
                    if (v > best[r]) { best[r] = v; bidx[r] = col; }
                }
            }
        }
    }

    // cross-lane argmax over the 16 columns held within each 16-lane half
#pragma unroll
    for (int r = 0; r < 8; ++r) {
        float v = best[r];
        int   idx = bidx[r];
#pragma unroll
        for (int off = 8; off >= 1; off >>= 1) {
            float ov = __shfl_xor(v, off, 16);
            int   oi = __shfl_xor(idx, off, 16);
            if (ov > v || (ov == v && oi < idx)) { v = ov; idx = oi; }
        }
        best[r] = v; bidx[r] = idx;
    }

    if (l15 == 0) {                                   // lanes 0 and 16 write
        const int rb = rowbase + hl * 8;
#pragma unroll
        for (int r = 0; r < 8; ++r) nearest[rb + r] = bidx[r];
    }
}

// ---------------- gather + decode: out = values[nearest] @ dec_w^T + dec_b ----
__global__ void __launch_bounds__(256)
kv_decode(const float* __restrict__ values, const int* __restrict__ nearest,
          const float* __restrict__ dec_w, const float* __restrict__ dec_b,
          float* __restrict__ out) {
    __shared__ float sw[NOUT * VDIM];                 // 40 KB
    __shared__ float sb[NOUT];

    const int tid = threadIdx.x;
#pragma unroll
    for (int j = 0; j < 10; ++j) {                    // 2560 float4 / 256 threads
        int c = tid + j * 256;
        *(float4*)&sw[c * 4] = *(const float4*)(dec_w + c * 4);
    }
    if (tid < NOUT) sb[tid] = dec_b[tid];
    __syncthreads();

    const int lane = tid & 31;
    const int wave = tid >> 5;
    const int row  = blockIdx.x * 8 + wave;
    const int n    = nearest[row];
    const float* vr = values + (size_t)n * VDIM;

    float p[NOUT];
#pragma unroll
    for (int o = 0; o < NOUT; ++o) p[o] = 0.f;

#pragma unroll
    for (int j = 0; j < 8; ++j) {
        const int vi = (lane + j * 32) * 4;
        float4 v = *(const float4*)(vr + vi);
#pragma unroll
        for (int o = 0; o < NOUT; ++o) {
            float4 w = *(const float4*)&sw[o * VDIM + vi];
            p[o] += v.x * w.x + v.y * w.y + v.z * w.z + v.w * w.w;
        }
    }
#pragma unroll
    for (int o = 0; o < NOUT; ++o)
#pragma unroll
        for (int off = 16; off >= 1; off >>= 1)
            p[o] += __shfl_xor(p[o], off, 32);

    if (lane == 0) {
#pragma unroll
        for (int o = 0; o < NOUT; ++o) out[(size_t)row * NOUT + o] = p[o] + sb[o];
    }
}

// ---------------- launch ----------------
extern "C" void kernel_launch(void* const* d_in, const int* in_sizes, int n_in,
                              void* d_out, int out_size, void* d_ws, size_t ws_size,
                              hipStream_t stream) {
    const float* x      = (const float*)d_in[0];
    const float* keys   = (const float*)d_in[1];
    const float* values = (const float*)d_in[2];
    const float* dec_w  = (const float*)d_in[3];
    const float* dec_b  = (const float*)d_in[4];
    float* out = (float*)d_out;

    unsigned short* xb = (unsigned short*)d_ws;                       // 32 MB
    unsigned short* kb = xb + (size_t)BATCH * KDIM;                   //  8 MB
    int* nearest       = (int*)(kb + (size_t)NKEY * KDIM);            // 64 KB

    const int nx8 = BATCH * KDIM / 8;   // 2,097,152
    const int nk8 = NKEY * KDIM / 8;    //   524,288
    cvt_f32_to_bf16<<<(nx8 + 255) / 256, 256, 0, stream>>>(x, xb, nx8);
    cvt_f32_to_bf16<<<(nk8 + 255) / 256, 256, 0, stream>>>(keys, kb, nk8);

    kv_gemm_argmax<<<BATCH / 128, 256, 0, stream>>>(xb, kb, nearest);

    kv_decode<<<BATCH / 8, 256, 0, stream>>>(values, nearest, dec_w, dec_b, out);
}
